// CombinatorialClassifier_11467562680599
// MI455X (gfx1250) — matrix operand order repaired and checked by reading, not compiled, not run on hardware
//
#include <hip/hip_runtime.h>
#include <hip/hip_bf16.h>
#include <stdint.h>

// Problem constants (from reference)
#define B_   256
#define P_   64
#define Q_   128
#define C_   1000
#define F_   2048

#define KSTEP      32
#define MTILE      32                   // b-rows per workgroup (2 WMMA tiles deep)
#define A_STRIDE   (F_ + 8)             // bf16 elems per padded LDS row (4112 B)
#define LG_STRIDE  (Q_ + 4)             // padded logits row (floats)

// dynamic-LDS layout (bytes)
#define SA_BYTES     (MTILE * A_STRIDE * 2)            // 131584
#define SLOGIT_OFF   SA_BYTES
#define SLOGIT_BYTES (MTILE * LG_STRIDE * 4)           // 16896
#define SRED_OFF     (SLOGIT_OFF + SLOGIT_BYTES)       // 148480
#define SRED_BYTES   (MTILE * 8 * 4)                   // 1024
#define SLSE_OFF     (SRED_OFF + SRED_BYTES)           // 149504
#define SMEM_TOTAL   (SLSE_OFF + MTILE * 4)            // 149632

typedef __attribute__((ext_vector_type(16))) __bf16 v16bf;
typedef __attribute__((ext_vector_type(8)))  float  v8f;

union BF16x16 { v16bf v; uint32_t u[8]; };

// fp32 -> packed bf16 pair, round-to-nearest-even (used only in one-time staging)
__device__ __forceinline__ uint32_t pk_bf16_rne(float lo, float hi) {
  uint32_t a = __builtin_bit_cast(uint32_t, lo);
  uint32_t b = __builtin_bit_cast(uint32_t, hi);
  a += 0x7FFFu + ((a >> 16) & 1u);
  b += 0x7FFFu + ((b >> 16) & 1u);
  return (a >> 16) | (b & 0xFFFF0000u);
}

// fp32 -> packed bf16 pair, truncating: ONE v_perm_b32.
// {S0,S1} byte pool: S0 = hi word (bytes 7..4), S1 = lo word (bytes 3..0).
// dst = {hi.b3, hi.b2, lo.b3, lo.b2}  -> selector 0x07060302
__device__ __forceinline__ uint32_t pk_bf16_fast(float lo, float hi) {
  return __builtin_amdgcn_perm(__builtin_bit_cast(uint32_t, hi),
                               __builtin_bit_cast(uint32_t, lo),
                               0x07060302u);
}

__global__ __launch_bounds__(256, 1) void cc_fused_kernel(
    const float* __restrict__ x,      // [B,F]
    const float* __restrict__ W,      // [P,Q,F]
    const float* __restrict__ bias,   // [P,Q]
    const int*   __restrict__ pidx,   // [P,C]
    float* __restrict__ out)          // [B,P,C]
{
  extern __shared__ __align__(16) char smem[];
  uint16_t* sA     = (uint16_t*)smem;                  // x tile as bf16 [MTILE][A_STRIDE]
  float*    sLogit = (float*)(smem + SLOGIT_OFF);      // [MTILE][LG_STRIDE]
  float*    sRed   = (float*)(smem + SRED_OFF);        // [MTILE][8]
  float*    sLse   = (float*)(smem + SLSE_OFF);        // [MTILE]

  const int tid  = threadIdx.x;
  const int wave = tid >> 5;          // 0..7 -> q tile
  const int lane = tid & 31;
  const int half = lane >> 4;         // lane group within wave
  const int l15  = lane & 15;

  const int b0 = blockIdx.x * MTILE;  // b-tile origin
  const int p  = blockIdx.y;          // partition

  // ---- Stage x tile [MTILE x F] into LDS as packed bf16 (coalesced float2 reads) ----
  {
    uint32_t* sAu = (uint32_t*)sA;
    for (int i = tid; i < MTILE * (F_ / 2); i += 256) {
      const int m  = i >> 10;                 // F_/2 == 1024
      const int kp = i & 1023;                // bf16 pair index within row
      const float2 v = ((const float2*)(x + (size_t)(b0 + m) * F_))[kp];
      sAu[m * (A_STRIDE / 2) + kp] = pk_bf16_rne(v.x, v.y);
    }
  }
  __syncthreads();

  // ---- GEMM: each wave -> two 16(b) x 16(q) tiles sharing one B tile, K = 2048 ----
  const int q0 = wave * 16;
  const float* wrow = W + ((size_t)p * Q_ + (q0 + l15)) * F_;          // B column
  // A row pointers per ISA 16-bit A 16x32 layout: lane l15 = row M, 'half' picks K sub-block
  const uint16_t* aRow0 = sA + l15        * A_STRIDE + half * 8;
  const uint16_t* aRow1 = sA + (16 + l15) * A_STRIDE + half * 8;

  v8f acc0 = {};
  v8f acc1 = {};
  #pragma unroll 2
  for (int k = 0; k < F_; k += KSTEP) {
    // B (32x16 layout): lane holds column q0+l15; u[0..3] = K k+half*8+{0..7}; u[4..7] = K k+16+half*8+{0..7}
    BF16x16 bm;
    const float4* wp = (const float4*)(wrow + k + half * 8);
    const float4 w0 = wp[0], w1 = wp[1], w2 = wp[4], w3 = wp[5];
    bm.u[0] = pk_bf16_fast(w0.x, w0.y); bm.u[1] = pk_bf16_fast(w0.z, w0.w);
    bm.u[2] = pk_bf16_fast(w1.x, w1.y); bm.u[3] = pk_bf16_fast(w1.z, w1.w);
    bm.u[4] = pk_bf16_fast(w2.x, w2.y); bm.u[5] = pk_bf16_fast(w2.z, w2.w);
    bm.u[6] = pk_bf16_fast(w3.x, w3.y); bm.u[7] = pk_bf16_fast(w3.z, w3.w);

    // A tiles (16-bit 16x32): u[0..3] = K k+half*8+{0..7}; u[4..7] = K k+16+half*8+{0..7}
    BF16x16 a0, a1;
    {
      const uint4 t0 = *(const uint4*)(aRow0 + k);
      const uint4 t1 = *(const uint4*)(aRow0 + k + 16);
      a0.u[0] = t0.x; a0.u[1] = t0.y; a0.u[2] = t0.z; a0.u[3] = t0.w;
      a0.u[4] = t1.x; a0.u[5] = t1.y; a0.u[6] = t1.z; a0.u[7] = t1.w;
    }
    {
      const uint4 t0 = *(const uint4*)(aRow1 + k);
      const uint4 t1 = *(const uint4*)(aRow1 + k + 16);
      a1.u[0] = t0.x; a1.u[1] = t0.y; a1.u[2] = t0.z; a1.u[3] = t0.w;
      a1.u[4] = t1.x; a1.u[5] = t1.y; a1.u[6] = t1.z; a1.u[7] = t1.w;
    }

    acc0 = __builtin_amdgcn_wmma_f32_16x16x32_bf16(
        false, a0.v, false, bm.v, (short)0, acc0, false, false);
    acc1 = __builtin_amdgcn_wmma_f32_16x16x32_bf16(
        false, a1.v, false, bm.v, (short)0, acc1, false, false);
  }

  // ---- bias + spill logits tiles to LDS (C/D layout: VGPR r -> M = r + half*8, N = l15) ----
  const float bv = bias[p * Q_ + q0 + l15];
  #pragma unroll
  for (int r = 0; r < 8; ++r) {
    const int m = r + half * 8;
    sLogit[m        * LG_STRIDE + q0 + l15] = acc0[r] + bv;
    sLogit[(16 + m) * LG_STRIDE + q0 + l15] = acc1[r] + bv;
  }
  __syncthreads();

  // ---- log-softmax over q (128) per row: 32 rows x 8 segments = 256 threads ----
  {
    const int row = tid >> 3, seg = tid & 7;
    const float* rp = sLogit + row * LG_STRIDE + seg * 16;
    float mx = rp[0];
    #pragma unroll
    for (int j = 1; j < 16; ++j) mx = fmaxf(mx, rp[j]);
    sRed[row * 8 + seg] = mx;
  }
  __syncthreads();
  if (tid < MTILE) {
    float mx = sRed[tid * 8];
    #pragma unroll
    for (int j = 1; j < 8; ++j) mx = fmaxf(mx, sRed[tid * 8 + j]);
    sLse[tid] = mx;                                 // row max (temporarily)
  }
  __syncthreads();
  {
    const int row = tid >> 3, seg = tid & 7;
    const float mx = sLse[row];
    const float* rp = sLogit + row * LG_STRIDE + seg * 16;
    float s = 0.f;
    #pragma unroll
    for (int j = 0; j < 16; ++j) s += __expf(rp[j] - mx);
    sRed[row * 8 + seg] = s;
  }
  __syncthreads();
  if (tid < MTILE) {
    float s = 0.f;
    #pragma unroll
    for (int j = 0; j < 8; ++j) s += sRed[tid * 8 + j];
    sLse[tid] += __logf(s);                          // lse = max + log(sum)
  }
  __syncthreads();

  // ---- gather 1000 classes per row: out[b0+m, p, c] = logit[m, idx] - lse[m] ----
  const size_t outBase = ((size_t)b0 * P_ + p) * C_;
  for (int i = tid; i < MTILE * C_; i += 256) {
    const int m = i / C_;
    const int c = i - m * C_;
    const int q = pidx[p * C_ + c];
    out[outBase + (size_t)m * (P_ * C_) + c] = sLogit[m * LG_STRIDE + q] - sLse[m];
  }
}

extern "C" void kernel_launch(void* const* d_in, const int* in_sizes, int n_in,
                              void* d_out, int out_size, void* d_ws, size_t ws_size,
                              hipStream_t stream) {
  (void)in_sizes; (void)n_in; (void)out_size; (void)d_ws; (void)ws_size;
  const float* x    = (const float*)d_in[0];
  const float* W    = (const float*)d_in[1];
  const float* bias = (const float*)d_in[2];
  const int*   pidx = (const int*)d_in[3];
  float* out = (float*)d_out;

  dim3 grid(B_ / MTILE, P_);       // 8 b-tiles x 64 partitions = 512 WGs
  cc_fused_kernel<<<grid, 256, SMEM_TOTAL, stream>>>(x, W, bias, pidx, out);
}
